// LSTMModel_41515153883281
// MI455X (gfx1250) — matrix operand order repaired and checked by reading, not compiled, not run on hardware
//
#include <hip/hip_runtime.h>
#include <hip/hip_bf16.h>

// ---------------------------------------------------------------------------
// LSTM (B=64, T=512, I=256, H=1024) for MI455X / gfx1250, bf16 WMMA + f32 acc.
// ---------------------------------------------------------------------------

typedef __attribute__((ext_vector_type(16))) __bf16 v16bf;
typedef __attribute__((ext_vector_type(8)))  __bf16 bf16x8;
typedef __attribute__((ext_vector_type(8)))  float  v8f;

#define B_  64
#define T_  512
#define I_  256
#define H_  1024
#define G4  4096   // 4*H
#define KC  1280   // H + I (concatenated K for fused [h|x] @ [W_hh|W_ih]^T)

union Frag { v16bf v; bf16x8 h[2]; };

__device__ __forceinline__ float fsig(float x) {
    return 1.0f / (1.0f + __expf(-x));
}
__device__ __forceinline__ float ftanh(float x) {
    // tanh(x) = 1 - 2/(exp(2x)+1); saturates correctly at +/-inf
    float e = __expf(2.0f * x);
    return 1.0f - 2.0f / (e + 1.0f);
}

// ---------------------------------------------------------------------------
// One-time packing: Wcat[p][k], p = 4*j + gate (gate-interleaved rows) so a
// contiguous 64-column tile holds all 4 gates for 16 hidden units.
//   k <  H : W_hh[row][k]
//   k >= H : W_ih[row][k-H]
// Also folds bias: bsum[p] = b_ih[row] + b_hh[row].
// ---------------------------------------------------------------------------
__global__ void pack_weights(const float* __restrict__ W_ih,
                             const float* __restrict__ W_hh,
                             const float* __restrict__ b_ih,
                             const float* __restrict__ b_hh,
                             __bf16* __restrict__ Wcat,
                             float* __restrict__ bsum) {
    int p = blockIdx.y;                          // 0..4095 packed row
    int k = blockIdx.x * 128 + threadIdx.x;      // 0..1279
    int gidx = p & 3;
    int j    = p >> 2;
    int r    = gidx * H_ + j;                    // original row in [4H, *]
    float v = (k < H_) ? W_hh[(size_t)r * H_ + k]
                       : W_ih[(size_t)r * I_ + (k - H_)];
    Wcat[(size_t)p * KC + k] = (__bf16)v;
    if (k == 0) bsum[p] = b_ih[r] + b_hh[r];
}

__global__ void cvt_x(const float* __restrict__ x, __bf16* __restrict__ xbf, int n) {
    int i = blockIdx.x * 256 + threadIdx.x;
    if (i < n) xbf[i] = (__bf16)x[i];
}

__global__ void init_state(__bf16* __restrict__ h0, float* __restrict__ c0) {
    int i = blockIdx.x * 256 + threadIdx.x;      // 65536 threads
    h0[i] = (__bf16)0.0f;
    c0[i] = 0.0f;
}

// ---------------------------------------------------------------------------
// One recurrence step: gates = [h|x_t] @ Wcat^T  (M=64, N=4096, K=1280)
// Grid: 64 WGs x 128 threads (4 waves). WG tile = 64M x 64N; wave = 16M x 64N.
// Epilogue: activations, c/h update, partial dot with w_fc.
// WG0 additionally reduces the PREVIOUS step's partials into out[:, t-1]
// (deterministic serial sum; part buffers are double-buffered by t parity).
// ---------------------------------------------------------------------------
__global__ void __launch_bounds__(128) lstm_step(
    const __bf16* __restrict__ Wcat, const float* __restrict__ bsum,
    const __bf16* __restrict__ xbf,  const __bf16* __restrict__ h_in,
    __bf16* __restrict__ h_out,      float* __restrict__ c_st,
    const float* __restrict__ w_fc,  const float* __restrict__ b_fc,
    const float* __restrict__ part_prev, float* __restrict__ part_cur,
    float* __restrict__ out, int t)
{
    __shared__ float lds_g[64][68];  // 64x64 gate tile, padded stride

    const int tid  = threadIdx.x;
    const int wg   = blockIdx.x;     // 0..63 -> hidden units [wg*16, wg*16+16)
    const int lane = tid & 31;
    const int wid  = tid >> 5;       // wave 0..3
    const int half = lane >> 4;      // 0|1
    const int lmod = lane & 15;
    const int m0   = wid * 16;       // batch-row base of this wave
    const int nb   = wg * 64;        // packed gate-column base of this WG

    // ---- deterministic reduction of previous step's output partials ----
    if (wg == 0 && t > 0 && tid < 64) {
        float s = b_fc[0];
        #pragma unroll 8
        for (int w = 0; w < 64; ++w) s += part_prev[w * 64 + tid];
        out[tid * T_ + (t - 1)] = s;
    }

    // ---- accumulators: 4 WMMA tiles of 16x16 ----
    v8f acc[4];
    #pragma unroll
    for (int i = 0; i < 4; ++i)
        #pragma unroll
        for (int e = 0; e < 8; ++e) acc[i][e] = 0.0f;

    // B fragment row pointers (packed weight rows, K-contiguous).
    // gfx1250 16-bit B layout: lane holds col = lane%16, K = half*8 + {0..7} and half*8 + {16..23}.
    const __bf16* wrow[4];
    #pragma unroll
    for (int tI = 0; tI < 4; ++tI)
        wrow[tI] = Wcat + (size_t)(nb + tI * 16 + lmod) * KC + half * 8;

    // A fragment (h part). gfx1250 16-bit A layout:
    // lane holds row = lane%16, K = half*8 + {0..7} and half*8 + {16..23}.
    const __bf16* arow = h_in + (m0 + lmod) * H_ + half * 8;

    // ---- K phase 1: recurrent part, K = 0..1023 over h_in ----
    #pragma unroll 4
    for (int kb = 0; kb < H_; kb += 32) {
        Frag A;
        A.h[0] = *(const bf16x8*)(arow + kb);
        A.h[1] = *(const bf16x8*)(arow + kb + 16);
        #pragma unroll
        for (int tI = 0; tI < 4; ++tI) {
            Frag Bf;
            Bf.h[0] = *(const bf16x8*)(wrow[tI] + kb);
            Bf.h[1] = *(const bf16x8*)(wrow[tI] + kb + 16);
            acc[tI] = __builtin_amdgcn_wmma_f32_16x16x32_bf16(
                false, A.v, false, Bf.v, (short)0, acc[tI], false, false);
        }
    }

    // ---- K phase 2: input part, K = 1024..1279 over x_t ----
    const __bf16* xrow = xbf + ((size_t)(m0 + lmod) * T_ + t) * I_ + half * 8;
    #pragma unroll 4
    for (int kb = 0; kb < I_; kb += 32) {
        Frag A;
        A.h[0] = *(const bf16x8*)(xrow + kb);
        A.h[1] = *(const bf16x8*)(xrow + kb + 16);
        #pragma unroll
        for (int tI = 0; tI < 4; ++tI) {
            Frag Bf;
            Bf.h[0] = *(const bf16x8*)(wrow[tI] + H_ + kb);
            Bf.h[1] = *(const bf16x8*)(wrow[tI] + H_ + kb + 16);
            acc[tI] = __builtin_amdgcn_wmma_f32_16x16x32_bf16(
                false, A.v, false, Bf.v, (short)0, acc[tI], false, false);
        }
    }

    // ---- spill gate tile to LDS (C layout: VGPR r -> M = r + 8*half, N = lmod) ----
    #pragma unroll
    for (int tI = 0; tI < 4; ++tI)
        #pragma unroll
        for (int r = 0; r < 8; ++r)
            lds_g[m0 + r + half * 8][tI * 16 + lmod] = acc[tI][r];
    __syncthreads();

    // ---- elementwise epilogue: 64 rows x 16 hidden units per WG ----
    #pragma unroll
    for (int s = 0; s < 8; ++s) {
        int idx = tid + s * 128;         // 0..1023
        int m   = idx >> 4;              // batch row 0..63
        int jl  = idx & 15;              // local hidden unit 0..15 (== tid&15)
        float gi = lds_g[m][4 * jl + 0] + bsum[nb + 4 * jl + 0];
        float gf = lds_g[m][4 * jl + 1] + bsum[nb + 4 * jl + 1];
        float gg = lds_g[m][4 * jl + 2] + bsum[nb + 4 * jl + 2];
        float go = lds_g[m][4 * jl + 3] + bsum[nb + 4 * jl + 3];
        float i_ = fsig(gi), f_ = fsig(gf), g_ = ftanh(gg), o_ = fsig(go);

        int jg = wg * 16 + jl;           // global hidden unit
        int ci = m * H_ + jg;
        float cn = f_ * c_st[ci] + i_ * g_;
        c_st[ci] = cn;
        float h = o_ * ftanh(cn);
        h_out[ci] = (__bf16)h;

        // per-(wg,m) partial of h . w_fc over this WG's 16 hidden units
        float pv = h * w_fc[jg];
        pv += __shfl_xor(pv, 1, 32);
        pv += __shfl_xor(pv, 2, 32);
        pv += __shfl_xor(pv, 4, 32);
        pv += __shfl_xor(pv, 8, 32);
        if (jl == 0) part_cur[wg * 64 + m] = pv;
    }
}

__global__ void finalize_out(const float* __restrict__ part,
                             const float* __restrict__ b_fc,
                             float* __restrict__ out, int t) {
    int m = threadIdx.x;
    if (m < 64) {
        float s = b_fc[0];
        #pragma unroll 8
        for (int w = 0; w < 64; ++w) s += part[w * 64 + m];
        out[m * T_ + t] = s;
    }
}

// ---------------------------------------------------------------------------
extern "C" void kernel_launch(void* const* d_in, const int* in_sizes, int n_in,
                              void* d_out, int out_size, void* d_ws, size_t ws_size,
                              hipStream_t stream) {
    (void)in_sizes; (void)n_in; (void)out_size; (void)ws_size;

    const float* x    = (const float*)d_in[0];   // [64,512,256]
    const float* W_ih = (const float*)d_in[1];   // [4096,256]
    const float* W_hh = (const float*)d_in[2];   // [4096,1024]
    const float* b_ih = (const float*)d_in[3];   // [4096]
    const float* b_hh = (const float*)d_in[4];   // [4096]
    const float* W_fc = (const float*)d_in[5];   // [1,1024]
    const float* b_fc = (const float*)d_in[6];   // [1]
    float* out = (float*)d_out;                  // [64,512]

    // workspace carve-up (256B aligned)
    char* ws = (char*)d_ws;
    size_t off = 0;
    auto carve = [&](size_t bytes) -> char* {
        char* p = ws + off;
        off = (off + bytes + 255) & ~(size_t)255;
        return p;
    };
    __bf16* Wcat  = (__bf16*)carve((size_t)G4 * KC * 2);       // 10.5 MB
    float*  bsum  = (float*) carve((size_t)G4 * 4);
    __bf16* xbf   = (__bf16*)carve((size_t)B_ * T_ * I_ * 2);  // 16 MB
    __bf16* hA    = (__bf16*)carve((size_t)B_ * H_ * 2);
    __bf16* hB    = (__bf16*)carve((size_t)B_ * H_ * 2);
    float*  c_st  = (float*) carve((size_t)B_ * H_ * 4);
    float*  partA = (float*) carve((size_t)64 * 64 * 4);
    float*  partB = (float*) carve((size_t)64 * 64 * 4);

    // one-time packing / conversion / init
    pack_weights<<<dim3(KC / 128, G4), 128, 0, stream>>>(W_ih, W_hh, b_ih, b_hh, Wcat, bsum);
    cvt_x<<<(B_ * T_ * I_ + 255) / 256, 256, 0, stream>>>(x, xbf, B_ * T_ * I_);
    init_state<<<(B_ * H_) / 256, 256, 0, stream>>>(hA, c_st);

    // sequential recurrence
    for (int t = 0; t < T_; ++t) {
        const __bf16* hin = (t & 1) ? hB : hA;
        __bf16* hout      = (t & 1) ? hA : hB;
        float* pprev      = (t & 1) ? partA : partB;  // written by step t-1
        float* pcur       = (t & 1) ? partB : partA;  // written by step t
        lstm_step<<<64, 128, 0, stream>>>(Wcat, bsum, xbf, hin, hout, c_st,
                                          W_fc, b_fc, pprev, pcur, out, t);
    }
    // out[:, 511] from the last step's partials (t=511 wrote partB)
    finalize_out<<<1, 64, 0, stream>>>(partB, b_fc, out, 511);
}
